// ChildSumTreeLSTM_22368189678451
// MI455X (gfx1250) — matrix operand order repaired and checked by reading, not compiled, not run on hardware
//
#include <hip/hip_runtime.h>
#include <hip/hip_bf16.h>

// ---------------------------------------------------------------------------
// Child-Sum TreeLSTM on gfx1250 (MI455X).
// Tree is static: 4-ary, 16384 leaves, levels contiguous in node order.
// Strategy: bf16 WMMA (v_wmma_f32_16x16x32_bf16) for every GEMM, f32 gates.
// GEMM: one wave owns a 32(M) x 64(N) tile -> 8 WMMAs per 32-deep K step with
// 12 b128 loads (1.5 loads/WMMA). A-row addressing is branchless (clamped):
// padded A rows only affect padded D rows, which are masked on store.
// ---------------------------------------------------------------------------

typedef __bf16 bf16;
typedef __attribute__((ext_vector_type(16))) __bf16 v16bf;
typedef __attribute__((ext_vector_type(8)))  __bf16 v8bf;
typedef __attribute__((ext_vector_type(8)))  float  v8f;

#define KDIM   512
#define NNODES 21845

__device__ __forceinline__ float sigf(float x) {
    return 1.0f / (1.0f + __expf(-x));
}

__device__ __forceinline__ v16bf cat16(v8bf lo, v8bf hi) {
    return __builtin_shufflevector(lo, hi, 0,1,2,3,4,5,6,7,8,9,10,11,12,13,14,15);
}

// ---------------------------------------------------------------------------
// C[M x Ncols] = A[M x 512](bf16) @ W + bias, with Bt[Ncols x 512] = W^T (bf16)
// Fragment layouts per CDNA5 ISA 7.12.2 (16-bit A 16x32, B 32x16, f32 C/D):
//   A frag : lane row = tileRow + (lane&15); half-wave selects K sub-ranges
//            {k..k+7, k+16..k+23} (lanes 0-15) / {k+8..k+15, k+24..k+31}
//   B frag : lane col = tileCol + (lane&15); same K sub-ranges as A frag
//            {k..k+7, k+16..k+23} / {k+8..k+15, k+24..k+31} in transposed weight
//   D frag : col = lane&15, row = 8*half + r for VGPR r
// ---------------------------------------------------------------------------
__global__ __launch_bounds__(128)
void gemm_bf16_wmma(const bf16* __restrict__ A, const bf16* __restrict__ Bt,
                    const float* __restrict__ bias, float* __restrict__ C,
                    int Mrows, int Ncols)
{
    const int lane = threadIdx.x & 31;
    const int wave = threadIdx.x >> 5;
    const int wid  = blockIdx.x * 4 + wave;

    const int MT = (Mrows + 31) >> 5;        // 32-row M tiles
    const int NT = Ncols >> 6;               // 64-col N tiles
    if (wid >= MT * NT) return;
    const int tm = wid / NT;
    const int tn = wid % NT;

    const int half = lane >> 4;              // 0/1 half-wave
    const int l16  = lane & 15;

    // Branchless clamped A row addressing (no zero-fill, no exec branches).
    const int mLim = Mrows - 1;
    int r0 = tm * 32 + l16;        r0 = r0 < mLim ? r0 : mLim;
    int r1 = tm * 32 + 16 + l16;   r1 = r1 < mLim ? r1 : mLim;
    const bf16* aBase0 = A + (size_t)r0 * KDIM;
    const bf16* aBase1 = A + (size_t)r1 * KDIM;

    const int nBase = tn * 64;
    const bf16* bBase0 = Bt + (size_t)(nBase +  0 + l16) * KDIM;
    const bf16* bBase1 = Bt + (size_t)(nBase + 16 + l16) * KDIM;
    const bf16* bBase2 = Bt + (size_t)(nBase + 32 + l16) * KDIM;
    const bf16* bBase3 = Bt + (size_t)(nBase + 48 + l16) * KDIM;

    v8f acc00 = {}; v8f acc01 = {}; v8f acc02 = {}; v8f acc03 = {};
    v8f acc10 = {}; v8f acc11 = {}; v8f acc12 = {}; v8f acc13 = {};

    for (int kk = 0; kk < KDIM; kk += 32) {
        const int kA = kk + 8 * half;
        v16bf a0 = cat16(*(const v8bf*)(aBase0 + kA), *(const v8bf*)(aBase0 + kA + 16));
        v16bf a1 = cat16(*(const v8bf*)(aBase1 + kA), *(const v8bf*)(aBase1 + kA + 16));

        const int kB = kk + 8 * half;
        v16bf b0 = cat16(*(const v8bf*)(bBase0 + kB), *(const v8bf*)(bBase0 + kB + 16));
        v16bf b1 = cat16(*(const v8bf*)(bBase1 + kB), *(const v8bf*)(bBase1 + kB + 16));
        v16bf b2 = cat16(*(const v8bf*)(bBase2 + kB), *(const v8bf*)(bBase2 + kB + 16));
        v16bf b3 = cat16(*(const v8bf*)(bBase3 + kB), *(const v8bf*)(bBase3 + kB + 16));

        acc00 = __builtin_amdgcn_wmma_f32_16x16x32_bf16(false, a0, false, b0, (short)0, acc00, false, false);
        acc01 = __builtin_amdgcn_wmma_f32_16x16x32_bf16(false, a0, false, b1, (short)0, acc01, false, false);
        acc02 = __builtin_amdgcn_wmma_f32_16x16x32_bf16(false, a0, false, b2, (short)0, acc02, false, false);
        acc03 = __builtin_amdgcn_wmma_f32_16x16x32_bf16(false, a0, false, b3, (short)0, acc03, false, false);
        acc10 = __builtin_amdgcn_wmma_f32_16x16x32_bf16(false, a1, false, b0, (short)0, acc10, false, false);
        acc11 = __builtin_amdgcn_wmma_f32_16x16x32_bf16(false, a1, false, b1, (short)0, acc11, false, false);
        acc12 = __builtin_amdgcn_wmma_f32_16x16x32_bf16(false, a1, false, b2, (short)0, acc12, false, false);
        acc13 = __builtin_amdgcn_wmma_f32_16x16x32_bf16(false, a1, false, b3, (short)0, acc13, false, false);
    }

    const float bv0 = bias[nBase      + l16];
    const float bv1 = bias[nBase + 16 + l16];
    const float bv2 = bias[nBase + 32 + l16];
    const float bv3 = bias[nBase + 48 + l16];

    #pragma unroll
    for (int r = 0; r < 8; ++r) {
        const int row0 = tm * 32 + 8 * half + r;
        const int row1 = row0 + 16;
        if (row0 < Mrows) {
            float* cRow = C + (size_t)row0 * Ncols + nBase + l16;
            cRow[0]  = acc00[r] + bv0;
            cRow[16] = acc01[r] + bv1;
            cRow[32] = acc02[r] + bv2;
            cRow[48] = acc03[r] + bv3;
        }
        if (row1 < Mrows) {
            float* cRow = C + (size_t)row1 * Ncols + nBase + l16;
            cRow[0]  = acc10[r] + bv0;
            cRow[16] = acc11[r] + bv1;
            cRow[32] = acc12[r] + bv2;
            cRow[48] = acc13[r] + bv3;
        }
    }
}

// ---------------------------------------------------------------------------
// Conversion helpers
// ---------------------------------------------------------------------------
__global__ __launch_bounds__(256)
void cvt_f32_to_bf16(const float* __restrict__ src, bf16* __restrict__ dst, int total)
{
    int t = blockIdx.x * blockDim.x + threadIdx.x;
    if (t < total) dst[t] = (bf16)src[t];
}

// Wt[n*512 + k] = (bf16) W[k*Ncols + n]  (transpose so B-fragments are contiguous)
__global__ __launch_bounds__(256)
void cvt_transpose_w(const float* __restrict__ W, bf16* __restrict__ Wt, int Ncols)
{
    int t = blockIdx.x * blockDim.x + threadIdx.x;
    int total = KDIM * Ncols;
    if (t >= total) return;
    int n = t / KDIM;
    int k = t - n * KDIM;
    Wt[t] = (bf16)W[(size_t)k * Ncols + n];
}

// ---------------------------------------------------------------------------
// Per-level elementwise kernels
// ---------------------------------------------------------------------------

// Leaves: children are zero -> iou = xi + b_iouh, c = i*u, h = o*tanh(c)
__global__ __launch_bounds__(256)
void leaf_gate_k(const float* __restrict__ x_iou, const float* __restrict__ b_iouh,
                 float* __restrict__ h_all, float* __restrict__ c_all,
                 bf16* __restrict__ hb_all)
{
    const int total = 16384 * 512;
    int t = blockIdx.x * blockDim.x + threadIdx.x;
    if (t >= total) return;
    int j = t >> 9, d = t & 511;
    size_t rb = (size_t)j * 1536;
    float gi = sigf (x_iou[rb +        d] + b_iouh[d]);
    float go = sigf (x_iou[rb +  512 + d] + b_iouh[512 + d]);
    float gu = tanhf(x_iou[rb + 1024 + d] + b_iouh[1024 + d]);
    float c = gi * gu;
    float h = go * tanhf(c);
    size_t o = (size_t)j * 512 + d;
    h_all[o] = h; c_all[o] = c; hb_all[o] = (bf16)h;
}

// h-sum over the 4 contiguous children rows, emitted directly as bf16
__global__ __launch_bounds__(256)
void hsum_k(const float* __restrict__ h_all, bf16* __restrict__ hsum_b,
            int offPrev, int nl)
{
    int t = blockIdx.x * blockDim.x + threadIdx.x;
    if (t >= nl * 512) return;
    int j = t >> 9, d = t & 511;
    const float* base = h_all + ((size_t)(offPrev + 4 * j)) * 512 + d;
    float s = base[0] + base[512] + base[1024] + base[1536];
    hsum_b[t] = (bf16)s;
}

// Internal nodes: combine precomputed x_iou/x_f with level GEMM results
__global__ __launch_bounds__(256)
void node_gate_k(const float* __restrict__ x_iou, const float* __restrict__ x_f,
                 const float* __restrict__ iou_part, const float* __restrict__ f_part,
                 float* __restrict__ h_all, float* __restrict__ c_all,
                 bf16* __restrict__ hb_all, int off, int offPrev, int nl)
{
    int t = blockIdx.x * blockDim.x + threadIdx.x;
    if (t >= nl * 512) return;
    int j = t >> 9, d = t & 511;
    size_t row = (size_t)(off + j);
    size_t rb  = row * 1536;
    size_t jb  = (size_t)j * 1536;
    float gi = sigf (x_iou[rb +        d] + iou_part[jb +        d]);  // b_iouh fused in GEMM
    float go = sigf (x_iou[rb +  512 + d] + iou_part[jb +  512 + d]);
    float gu = tanhf(x_iou[rb + 1024 + d] + iou_part[jb + 1024 + d]);
    float c  = gi * gu;
    float xf = x_f[row * 512 + d];
    #pragma unroll
    for (int k = 0; k < 4; ++k) {
        size_t ch = (size_t)(offPrev + 4 * j + k);
        float f = sigf(f_part[(size_t)(4 * j + k) * 512 + d] + xf);    // b_fh fused in GEMM
        c += f * c_all[ch * 512 + d];
    }
    float h = go * tanhf(c);
    size_t o = row * 512 + d;
    h_all[o] = h; c_all[o] = c; hb_all[o] = (bf16)h;
}

__global__ __launch_bounds__(256)
void copy_root_k(const float* __restrict__ h_all, float* __restrict__ out)
{
    int d = blockIdx.x * blockDim.x + threadIdx.x;
    if (d < 512) out[d] = h_all[(size_t)(NNODES - 1) * 512 + d];
}

// ---------------------------------------------------------------------------
// Host-side orchestration
// ---------------------------------------------------------------------------
static inline void launch_gemm(const bf16* A, const bf16* Bt, const float* bias,
                               float* C, int M, int Ncols, hipStream_t s)
{
    int MT = (M + 31) / 32;
    int NT = Ncols / 64;
    int waves = MT * NT;
    int blocks = (waves + 3) / 4;
    gemm_bf16_wmma<<<blocks, 128, 0, s>>>(A, Bt, bias, C, M, Ncols);
}

extern "C" void kernel_launch(void* const* d_in, const int* in_sizes, int n_in,
                              void* d_out, int out_size, void* d_ws, size_t ws_size,
                              hipStream_t stream)
{
    const float* x      = (const float*)d_in[0];
    // d_in[1] = children; tree is contiguous by construction, not needed.
    const float* W_ioux = (const float*)d_in[2];
    const float* b_ioux = (const float*)d_in[3];
    const float* W_iouh = (const float*)d_in[4];
    const float* b_iouh = (const float*)d_in[5];
    const float* W_fx   = (const float*)d_in[6];
    const float* b_fx   = (const float*)d_in[7];
    const float* W_fh   = (const float*)d_in[8];
    const float* b_fh   = (const float*)d_in[9];

    // ---- workspace carve-up (256B aligned slabs) ----
    size_t wo = 0;
    auto take = [&](size_t bytes) -> void* {
        void* p = (void*)((char*)d_ws + wo);
        wo += (bytes + 255) & ~(size_t)255;
        return p;
    };
    const size_t N = NNODES;
    float* x_iou    = (float*)take(N * 1536 * sizeof(float));
    float* x_f      = (float*)take(N * 512  * sizeof(float));
    float* h_all    = (float*)take(N * 512  * sizeof(float));
    float* c_all    = (float*)take(N * 512  * sizeof(float));
    bf16*  hb_all   = (bf16*) take(N * 512  * sizeof(bf16));
    bf16*  xb       = (bf16*) take(N * 512  * sizeof(bf16));
    bf16*  Wt_ioux  = (bf16*) take((size_t)512 * 1536 * sizeof(bf16));
    bf16*  Wt_iouh  = (bf16*) take((size_t)512 * 1536 * sizeof(bf16));
    bf16*  Wt_fx    = (bf16*) take((size_t)512 * 512  * sizeof(bf16));
    bf16*  Wt_fh    = (bf16*) take((size_t)512 * 512  * sizeof(bf16));
    bf16*  hsum_b   = (bf16*) take((size_t)4096 * 512 * sizeof(bf16));
    float* iou_part = (float*)take((size_t)4096 * 1536 * sizeof(float));
    float* f_part   = (float*)take((size_t)16384 * 512 * sizeof(float));
    (void)ws_size; (void)in_sizes; (void)n_in; (void)out_size;

    // ---- 1) convert inputs to bf16 (weights transposed) ----
    {
        int total = (int)(N * 512);
        cvt_f32_to_bf16<<<(total + 255) / 256, 256, 0, stream>>>(x, xb, total);
        cvt_transpose_w<<<(512 * 1536 + 255) / 256, 256, 0, stream>>>(W_ioux, Wt_ioux, 1536);
        cvt_transpose_w<<<(512 * 1536 + 255) / 256, 256, 0, stream>>>(W_iouh, Wt_iouh, 1536);
        cvt_transpose_w<<<(512 * 512  + 255) / 256, 256, 0, stream>>>(W_fx, Wt_fx, 512);
        cvt_transpose_w<<<(512 * 512  + 255) / 256, 256, 0, stream>>>(W_fh, Wt_fh, 512);
    }

    // ---- 2) big input GEMMs over all nodes ----
    launch_gemm(xb, Wt_ioux, b_ioux, x_iou, (int)N, 1536, stream);
    launch_gemm(xb, Wt_fx,   b_fx,   x_f,   (int)N, 512,  stream);

    // ---- 3) level 0 (leaves) ----
    leaf_gate_k<<<(16384 * 512 + 255) / 256, 256, 0, stream>>>(
        x_iou, b_iouh, h_all, c_all, hb_all);

    // ---- 4) internal levels (children of level l are level l-1, contiguous) ----
    static const int LVL_N[8]   = {16384, 4096, 1024, 256, 64, 16, 4, 1};
    static const int LVL_OFF[8] = {0, 16384, 20480, 21504, 21760, 21824, 21840, 21844};
    for (int l = 1; l < 8; ++l) {
        const int nl      = LVL_N[l];
        const int off     = LVL_OFF[l];
        const int offPrev = LVL_OFF[l - 1];

        hsum_k<<<(nl * 512 + 255) / 256, 256, 0, stream>>>(h_all, hsum_b, offPrev, nl);
        launch_gemm(hsum_b, Wt_iouh, b_iouh, iou_part, nl, 1536, stream);
        launch_gemm(hb_all + (size_t)offPrev * 512, Wt_fh, b_fh, f_part, 4 * nl, 512, stream);
        node_gate_k<<<(nl * 512 + 255) / 256, 256, 0, stream>>>(
            x_iou, x_f, iou_part, f_part, h_all, c_all, hb_all, off, offPrev, nl);
    }

    // ---- 5) emit root h ----
    copy_root_k<<<2, 256, 0, stream>>>(h_all, (float*)d_out);
}
